// Attention_42726334660908
// MI455X (gfx1250) — matrix operand order repaired and checked by reading, not compiled, not run on hardware
//
#include <hip/hip_runtime.h>

// CDNA5 / gfx1250 tied-row attention, bf16 WMMA pipeline.
// wave32; each GEMM wave computes a 32x64 C tile (8 wmma / K-step), with
// software-pipelined (double-buffered) fragment loads so WMMAs overlap VMEM.

typedef __attribute__((ext_vector_type(16))) __bf16 v16bf;
typedef __attribute__((ext_vector_type(8)))  __bf16 v8bf;
typedef __attribute__((ext_vector_type(8)))  float  v8f;

#define NHEADS 8
#define DIMH   64
#define SEQ    512
#define NB     128   // B
#define DMODEL 256
#define INNER  512   // NHEADS*DIMH
#define RTIE   64
#define BB     2     // NB / RTIE

__device__ __forceinline__ __bf16 f2bf(float f) {
  unsigned u = __builtin_bit_cast(unsigned, f);
  unsigned r = u + 0x7FFFu + ((u >> 16) & 1u);   // round-to-nearest-even
  unsigned short h = (unsigned short)(r >> 16);
  return __builtin_bit_cast(__bf16, h);
}

__device__ __forceinline__ v8f wmma_bf16(v16bf a, v16bf b, v8f c) {
  return __builtin_amdgcn_wmma_f32_16x16x32_bf16(false, a, false, b, (short)0, c,
                                                 false, false);
}

__device__ __forceinline__ v16bf mk16(v8bf lo, v8bf hi) {
  return __builtin_shufflevector(lo, hi, 0, 1, 2, 3, 4, 5, 6, 7,
                                 8, 9, 10, 11, 12, 13, 14, 15);
}

// A 16x32 bf16 fragment from row-major [M,K] source: two global_load_b128/lane.
__device__ __forceinline__ v16bf load_a(const __bf16* p, int ld) {
  int lane = threadIdx.x & 31;
  const __bf16* q = p + (size_t)(lane & 15) * ld + ((lane >> 4) << 3);
  v8bf lo = *(const v8bf*)(q);
  v8bf hi = *(const v8bf*)(q + 16);
  return mk16(lo, hi);
}

// B 32x16 bf16 fragment from N-major (transposed) [N,K] source: two b128/lane.
__device__ __forceinline__ v16bf load_bt(const __bf16* p, int ld) {
  int lane = threadIdx.x & 31;
  const __bf16* q = p + (size_t)(lane & 15) * ld + ((lane >> 4) << 3);
  v8bf lo = *(const v8bf*)(q);
  v8bf hi = *(const v8bf*)(q + 16);
  return mk16(lo, hi);
}

// C/D 16x16 f32: lane -> N = lane&15; VGPR i -> M = i + (lane>=16 ? 8 : 0)
__device__ __forceinline__ void store_f32(float* p, int ld, v8f c) {
  int lane = threadIdx.x & 31;
  int n = lane & 15;
  int mb = (lane >> 4) << 3;
#pragma unroll
  for (int i = 0; i < 8; ++i) p[(size_t)(mb + i) * ld + n] = c[i];
}

__device__ __forceinline__ void store_bf16(__bf16* p, int ld, v8f c) {
  int lane = threadIdx.x & 31;
  int n = lane & 15;
  int mb = (lane >> 4) << 3;
#pragma unroll
  for (int i = 0; i < 8; ++i) p[(size_t)(mb + i) * ld + n] = f2bf(c[i]);
}

// Transposed C store: (m,n) -> p[n*ld + m]; contiguous 16B store per lane.
__device__ __forceinline__ void store_bf16_T(__bf16* p, int ld, v8f c) {
  int lane = threadIdx.x & 31;
  int n = lane & 15;
  int mb = (lane >> 4) << 3;
  v8bf o;
#pragma unroll
  for (int i = 0; i < 8; ++i) o[i] = f2bf(c[i]);
  *(v8bf*)(p + (size_t)n * ld + mb) = o;
}

// ---- pipelined 32x64-tile helpers ----
struct Frag { v16bf a0, a1, b[4]; };

__device__ __forceinline__ Frag load_frags(const __bf16* pa, int lda,
                                           const __bf16* pbt, int ldb) {
  Frag f;
  f.a0 = load_a(pa, lda);
  f.a1 = load_a(pa + (size_t)16 * lda, lda);
#pragma unroll
  for (int t = 0; t < 4; ++t) f.b[t] = load_bt(pbt + (size_t)t * 16 * ldb, ldb);
  return f;
}

__device__ __forceinline__ void mma8(v8f acc[2][4], const Frag& f) {
#pragma unroll
  for (int t = 0; t < 4; ++t) {
    acc[0][t] = wmma_bf16(f.a0, f.b[t], acc[0][t]);
    acc[1][t] = wmma_bf16(f.a1, f.b[t], acc[1][t]);
  }
}

// ---- P1: f32 -> bf16 bulk convert (8 elems/thread) ----
__global__ void cvt_kernel(const float* __restrict__ in, __bf16* __restrict__ out) {
  size_t i = (size_t)(blockIdx.x * blockDim.x + threadIdx.x);
  const float4* p = (const float4*)in + i * 2;
  float4 a = p[0], b = p[1];
  v8bf o;
  o[0] = f2bf(a.x); o[1] = f2bf(a.y); o[2] = f2bf(a.z); o[3] = f2bf(a.w);
  o[4] = f2bf(b.x); o[5] = f2bf(b.y); o[6] = f2bf(b.z); o[7] = f2bf(b.w);
  ((v8bf*)out)[i] = o;
}

// ---- P2: weight transpose+convert: W[K,N] f32 -> WT[N,K] bf16 ----
__global__ void tconv_kernel(const float* __restrict__ W, __bf16* __restrict__ WT,
                             int K, int N) {
  int idx = blockIdx.x * blockDim.x + threadIdx.x;
  int n = idx / K, k = idx - n * K;
  WT[idx] = f2bf(W[(size_t)k * N + n]);
}

// ---- G1: row-major projection: C[65536,512] = xb[65536,256] @ WT[512,256]^T ----
__global__ void proj_row(const __bf16* __restrict__ A, const __bf16* __restrict__ BT,
                         __bf16* __restrict__ C) {
  int wave = (int)((blockIdx.x * blockDim.x + threadIdx.x) >> 5);
  int nt = wave & 7;            // 512/64
  int mt = wave >> 3;
  size_t m0 = (size_t)mt * 32;
  int n0 = nt * 64;
  const __bf16* pa = A + m0 * DMODEL;
  const __bf16* pb = BT + (size_t)n0 * DMODEL;
  v8f acc[2][4] = {};
  Frag f = load_frags(pa, DMODEL, pb, DMODEL);
  for (int k0 = 32; k0 < DMODEL; k0 += 32) {
    Frag nf = load_frags(pa + k0, DMODEL, pb + k0, DMODEL);
    mma8(acc, f);
    f = nf;
  }
  mma8(acc, f);
#pragma unroll
  for (int s = 0; s < 2; ++s)
#pragma unroll
    for (int t = 0; t < 4; ++t)
      store_bf16(C + (m0 + s * 16) * INNER + n0 + t * 16, INNER, acc[s][t]);
}

// ---- G2: V projection with transposed store: vT[g][feat][j] ----
__global__ void proj_vT(const __bf16* __restrict__ A, const __bf16* __restrict__ BT,
                        __bf16* __restrict__ vT) {
  int wave = (int)((blockIdx.x * blockDim.x + threadIdx.x) >> 5);
  int nt = wave & 7;
  int mt = wave >> 3;
  size_t m0 = (size_t)mt * 32;
  int n0 = nt * 64;
  int g = (int)(m0 >> 9);
  int jbase = (int)(m0 & 511);
  const __bf16* pa = A + m0 * DMODEL;
  const __bf16* pb = BT + (size_t)n0 * DMODEL;
  v8f acc[2][4] = {};
  Frag f = load_frags(pa, DMODEL, pb, DMODEL);
  for (int k0 = 32; k0 < DMODEL; k0 += 32) {
    Frag nf = load_frags(pa + k0, DMODEL, pb + k0, DMODEL);
    mma8(acc, f);
    f = nf;
  }
  mma8(acc, f);
#pragma unroll
  for (int s = 0; s < 2; ++s)
#pragma unroll
    for (int t = 0; t < 4; ++t)
      store_bf16_T(vT + ((size_t)g * INNER + n0 + t * 16) * SEQ + jbase + s * 16,
                   SEQ, acc[s][t]);
}

// ---- G3: dots[bh,i,j] = (1/64) * sum_{r,d} q[g,i,hd] k[g,j,hd];  K = r*d = 4096 ----
__device__ __forceinline__ void score_ptrs(const __bf16* qb, const __bf16* kb,
                                           int b, int h, int it, int jt, int kk,
                                           const __bf16** pa, const __bf16** pb) {
  int r = kk >> 6, d0 = kk & 63;
  int g = b * RTIE + r;
  *pa = qb + ((size_t)g * SEQ + it * 32) * INNER + h * DIMH + d0;
  *pb = kb + ((size_t)g * SEQ + jt * 64) * INNER + h * DIMH + d0;
}

__global__ void scores_kernel(const __bf16* __restrict__ qb,
                              const __bf16* __restrict__ kb,
                              float* __restrict__ sc) {
  int wave = (int)((blockIdx.x * blockDim.x + threadIdx.x) >> 5);
  int jt = wave & 7;            // j tile of 64
  int it = (wave >> 3) & 15;    // i tile of 32
  int bh = wave >> 7;           // 0..15
  int b = bh >> 3, h = bh & 7;
  v8f acc[2][4] = {};
  const __bf16 *pa, *pb;
  score_ptrs(qb, kb, b, h, it, jt, 0, &pa, &pb);
  Frag f = load_frags(pa, INNER, pb, INNER);
  for (int kk = 32; kk < RTIE * DIMH; kk += 32) {
    score_ptrs(qb, kb, b, h, it, jt, kk, &pa, &pb);
    Frag nf = load_frags(pa, INNER, pb, INNER);
    mma8(acc, f);
    f = nf;
  }
  mma8(acc, f);
  const float scale = 1.0f / 64.0f;   // dh^-0.5 * r^-0.5
#pragma unroll
  for (int s = 0; s < 2; ++s)
#pragma unroll
    for (int t = 0; t < 4; ++t) {
      v8f c = acc[s][t];
#pragma unroll
      for (int i = 0; i < 8; ++i) c[i] *= scale;
      store_f32(sc + ((size_t)bh * SEQ + it * 32 + s * 16) * SEQ + jt * 64 + t * 16,
                SEQ, c);
    }
}

// ---- S: row softmax over j (512), one wave per row, bf16 probs ----
__global__ void softmax_kernel(const float* __restrict__ sc, __bf16* __restrict__ pr) {
  int wave = (int)((blockIdx.x * blockDim.x + threadIdx.x) >> 5);
  int lane = threadIdx.x & 31;
  const float* row = sc + (size_t)wave * SEQ;
  __bf16* prow = pr + (size_t)wave * SEQ;
  float v[16];
  float mx = -3.4e38f;
#pragma unroll
  for (int i = 0; i < 16; ++i) { v[i] = row[lane + i * 32]; mx = fmaxf(mx, v[i]); }
#pragma unroll
  for (int off = 16; off > 0; off >>= 1) mx = fmaxf(mx, __shfl_xor(mx, off, 32));
  float s = 0.0f;
#pragma unroll
  for (int i = 0; i < 16; ++i) { v[i] = __expf(v[i] - mx); s += v[i]; }
#pragma unroll
  for (int off = 16; off > 0; off >>= 1) s += __shfl_xor(s, off, 32);
  float inv = 1.0f / s;
#pragma unroll
  for (int i = 0; i < 16; ++i) prow[lane + i * 32] = f2bf(v[i] * inv);
}

// ---- G4: out[g,i,hd] = sum_j probs[bh,i,j] * vT[g,hd,j];  K = 512 ----
__global__ void av_kernel(const __bf16* __restrict__ pr,
                          const __bf16* __restrict__ vT,
                          __bf16* __restrict__ aout) {
  int wave = (int)((blockIdx.x * blockDim.x + threadIdx.x) >> 5);
  int it = wave & 15;           // i tile of 32
  int brh = wave >> 4;          // 0..1023
  int h = brh & 7;
  int r = (brh >> 3) & 63;
  int b = brh >> 9;
  int g = b * RTIE + r;
  int bh = b * NHEADS + h;
  const __bf16* pa = pr + ((size_t)bh * SEQ + it * 32) * SEQ;
  const __bf16* pb = vT + ((size_t)g * INNER + h * DIMH) * SEQ;
  v8f acc[2][4] = {};
  Frag f = load_frags(pa, SEQ, pb, SEQ);
  for (int k0 = 32; k0 < SEQ; k0 += 32) {
    Frag nf = load_frags(pa + k0, SEQ, pb + k0, SEQ);
    mma8(acc, f);
    f = nf;
  }
  mma8(acc, f);
#pragma unroll
  for (int s = 0; s < 2; ++s)
#pragma unroll
    for (int t = 0; t < 4; ++t)
      store_bf16(aout + ((size_t)g * SEQ + it * 32 + s * 16) * INNER + h * DIMH + t * 16,
                 INNER, acc[s][t]);
}

// ---- G5: out[M,256] = aout[M,512] @ WoT[256,512]^T + bo (f32 out) ----
__global__ void out_kernel(const __bf16* __restrict__ aout,
                           const __bf16* __restrict__ WoT,
                           const float* __restrict__ bo, float* __restrict__ out) {
  int wave = (int)((blockIdx.x * blockDim.x + threadIdx.x) >> 5);
  int nt = wave & 3;            // 256 / 64
  int mt = wave >> 2;
  size_t m0 = (size_t)mt * 32;
  int n0 = nt * 64;
  const __bf16* pa = aout + m0 * INNER;
  const __bf16* pb = WoT + (size_t)n0 * INNER;
  v8f acc[2][4] = {};
  Frag f = load_frags(pa, INNER, pb, INNER);
  for (int k0 = 32; k0 < INNER; k0 += 32) {
    Frag nf = load_frags(pa + k0, INNER, pb + k0, INNER);
    mma8(acc, f);
    f = nf;
  }
  mma8(acc, f);
  int lane = threadIdx.x & 31;
  int ncol = lane & 15;
#pragma unroll
  for (int s = 0; s < 2; ++s)
#pragma unroll
    for (int t = 0; t < 4; ++t) {
      float bias = bo[n0 + t * 16 + ncol];
      v8f c = acc[s][t];
#pragma unroll
      for (int i = 0; i < 8; ++i) c[i] += bias;
      store_f32(out + (m0 + s * 16) * DMODEL + n0 + t * 16, DMODEL, c);
    }
}

extern "C" void kernel_launch(void* const* d_in, const int* in_sizes, int n_in,
                              void* d_out, int out_size, void* d_ws, size_t ws_size,
                              hipStream_t stream) {
  const float* x   = (const float*)d_in[0];   // [128, 512, 256]
  const float* Wq  = (const float*)d_in[1];   // [256, 512]
  const float* Wkv = (const float*)d_in[2];   // [256, 1024]
  const float* Wo  = (const float*)d_in[3];   // [512, 256]
  const float* bo  = (const float*)d_in[4];   // [256]
  float* out = (float*)d_out;

  char* ws = (char*)d_ws;
  __bf16* xb   = (__bf16*)(ws);                          //  32 MB [65536,256]
  __bf16* qb   = (__bf16*)(ws + ((size_t)32  << 20));    //  64 MB [65536,512]
  __bf16* kbuf = (__bf16*)(ws + ((size_t)96  << 20));    //  64 MB [65536,512]
  __bf16* vT   = (__bf16*)(ws + ((size_t)160 << 20));    //  64 MB [128,512,512]
  __bf16* aout = (__bf16*)(ws + ((size_t)224 << 20));    //  64 MB [65536,512]
  float*  sc   = (float*) (ws + ((size_t)288 << 20));    //  16 MB [16,512,512]
  __bf16* pr   = (__bf16*)(ws + ((size_t)304 << 20));    //   8 MB [16,512,512]
  __bf16* WqT  = (__bf16*)(ws + ((size_t)312 << 20));    // 256 KB [512,256]
  __bf16* WkvT = (__bf16*)(ws + ((size_t)313 << 20));    // 512 KB [1024,256]
  __bf16* WoT  = (__bf16*)(ws + ((size_t)314 << 20));    // 256 KB [256,512]

  const int M = NB * SEQ;       // 65536
  const int TPB = 256;          // 8 waves / block

  // P1/P2: operand preparation (bf16 + transposed weights)
  cvt_kernel<<<(M * DMODEL / 8) / TPB, TPB, 0, stream>>>(x, xb);
  tconv_kernel<<<(DMODEL * INNER) / TPB, TPB, 0, stream>>>(Wq, WqT, DMODEL, INNER);
  tconv_kernel<<<(DMODEL * 2 * INNER) / TPB, TPB, 0, stream>>>(Wkv, WkvT, DMODEL, 2 * INNER);
  tconv_kernel<<<(INNER * DMODEL) / TPB, TPB, 0, stream>>>(Wo, WoT, INNER, DMODEL);

  // G1: q and k projections
  {
    int waves = (M / 32) * (INNER / 64);                   // 16384
    proj_row<<<waves * 32 / TPB, TPB, 0, stream>>>(xb, WqT, qb);
    proj_row<<<waves * 32 / TPB, TPB, 0, stream>>>(xb, WkvT, kbuf);
  }
  // G2: v projection, stored transposed [g][feat][j]
  {
    int waves = (M / 32) * (INNER / 64);                   // 16384
    proj_vT<<<waves * 32 / TPB, TPB, 0, stream>>>(xb, WkvT + (size_t)INNER * DMODEL, vT);
  }
  // G3: tied scores, 16 x (512x512 GEMM, K=4096)
  {
    int waves = BB * NHEADS * (SEQ / 32) * (SEQ / 64);     // 2048
    scores_kernel<<<waves * 32 / TPB, TPB, 0, stream>>>(qb, kbuf, sc);
  }
  // S: softmax over 8192 rows
  {
    int waves = BB * NHEADS * SEQ;                         // 8192
    softmax_kernel<<<waves * 32 / TPB, TPB, 0, stream>>>(sc, pr);
  }
  // G4: attn @ V, 1024 x (512x64x512)
  {
    int waves = BB * RTIE * NHEADS * (SEQ / 32);           // 16384
    av_kernel<<<waves * 32 / TPB, TPB, 0, stream>>>(pr, vT, aout);
  }
  // G5: output projection + bias
  {
    int waves = (M / 32) * (DMODEL / 64);                  // 8192
    out_kernel<<<waves * 32 / TPB, TPB, 0, stream>>>(aout, WoT, bo, out);
  }
}
